// CAMIL_26431228739594
// MI455X (gfx1250) — matrix operand order repaired and checked by reading, not compiled, not run on hardware
//
#include <hip/hip_runtime.h>
#include <math.h>

// ---------------- problem constants ----------------
#define NI   20000      // instances (N)
#define DD   1024       // feature dim (D)
#define NPD  20224      // padded sequence length (front pad 224)
#define PADR 224
#define NH   8          // heads
#define DHD  64         // head dim
#define MLM  256        // landmarks
#define LSEG 79         // NPD / MLM
#define KCV  33         // residual conv taps
#define NE   320000     // edges
#define QKD  1536       // 3*NH*DHD
#define NIP  20096      // NI rounded up to 128

typedef _Float16 h8  __attribute__((ext_vector_type(8)));
typedef _Float16 h16 __attribute__((ext_vector_type(16)));
typedef float    f8  __attribute__((ext_vector_type(8)));
typedef int      camil_i4 __attribute__((vector_size(16)));

// ---------------- CDNA5 async global->LDS staging ----------------
#if defined(__HIP_DEVICE_COMPILE__) && __has_builtin(__builtin_amdgcn_global_load_async_to_lds_b128)
#define CAMIL_ASYNC 1
__device__ __forceinline__ void camil_cp16(_Float16* l, const _Float16* g) {
  __builtin_amdgcn_global_load_async_to_lds_b128(
      (__attribute__((address_space(1))) camil_i4*)g,
      (__attribute__((address_space(3))) camil_i4*)l, 0, 0);
}
__device__ __forceinline__ void camil_cp_wait4() {
#if __has_builtin(__builtin_amdgcn_s_wait_asynccnt)
  __builtin_amdgcn_s_wait_asynccnt(4);
#else
  asm volatile("s_wait_asynccnt 4" ::: "memory");
#endif
}
__device__ __forceinline__ void camil_cp_wait0() {
#if __has_builtin(__builtin_amdgcn_s_wait_asynccnt)
  __builtin_amdgcn_s_wait_asynccnt(0);
#else
  asm volatile("s_wait_asynccnt 0" ::: "memory");
#endif
}
#else
#define CAMIL_ASYNC 0
__device__ __forceinline__ void camil_cp16(_Float16* l, const _Float16* g) {
  *(h8*)l = *(const h8*)g;
}
__device__ __forceinline__ void camil_cp_wait4() {}
__device__ __forceinline__ void camil_cp_wait0() {}
#endif

// ---------------- block reductions (blockDim.x == 256) ----------------
__device__ __forceinline__ float camil_blk_max(float v, float* sm) {
  int t = threadIdx.x;
  sm[t] = v; __syncthreads();
  for (int s = 128; s > 0; s >>= 1) { if (t < s) sm[t] = fmaxf(sm[t], sm[t + s]); __syncthreads(); }
  float r = sm[0]; __syncthreads(); return r;
}
__device__ __forceinline__ float camil_blk_sum(float v, float* sm) {
  int t = threadIdx.x;
  sm[t] = v; __syncthreads();
  for (int s = 128; s > 0; s >>= 1) { if (t < s) sm[t] += sm[t + s]; __syncthreads(); }
  float r = sm[0]; __syncthreads(); return r;
}

// ---------------- WMMA GEMM (f16 operands, f32 accumulate) ----------------
// A: f16 [M,K] row-major (lda). B: f16 [N,K] K-contiguous (ldb).
// Operand buffers MUST be padded so rows up to ceil(M/128)*128 / ceil(N/128)*128
// are addressable (no bounds checks in staging; epilogue guards stores).
// out_mode: 0 = f32 C[row*ldc+col]; 1 = f16 C[row*ldc+col]; 2 = f16 C[col*ldc+row].
// K must be a multiple of 32. Batched via grid.z (element strides sA/sB/sC).
__global__ __launch_bounds__(256)
void camil_gemm_wmma_h(const _Float16* __restrict__ A, const _Float16* __restrict__ B,
                       void* __restrict__ Cv, const float* __restrict__ bias,
                       int M, int N, int K, int lda, int ldb, int ldc,
                       long sA, long sB, long sC, float alpha, int out_mode)
{
  __shared__ __align__(16) _Float16 SA[2][128 * 32];
  __shared__ __align__(16) _Float16 SB[2][128 * 32];
  A += (long)blockIdx.z * sA;
  B += (long)blockIdx.z * sB;
  const int m0 = blockIdx.y * 128, n0 = blockIdx.x * 128;
  const int tid = threadIdx.x, lane = tid & 31, wv = tid >> 5;
  const int wm = wv >> 1, wn = wv & 1;            // 4x2 waves -> 32x64 per wave
  const int hi = lane >> 4, rl = lane & 15;
  const int nk = K >> 5;

  f8 acc[2][4] = {};

  // issue async copies for k-tile kt into buffer kt&1 (4 ops per wave)
  auto stage = [&](int kt) {
    const int k0 = kt << 5;
    _Float16* as = SA[kt & 1];
    _Float16* bs = SB[kt & 1];
    #pragma unroll
    for (int it = 0; it < 2; ++it) {
      int f = (tid + it * 256) << 3;               // 8 halves (16B) per op
      int r = f >> 5, c = f & 31;
      camil_cp16(as + r * 32 + c, A + (long)(m0 + r) * lda + k0 + c);
      camil_cp16(bs + r * 32 + c, B + (long)(n0 + r) * ldb + k0 + c);
    }
  };

  stage(0);
  for (int kt = 0; kt < nk; ++kt) {
    if (kt + 1 < nk) { stage(kt + 1); camil_cp_wait4(); }
    else            { camil_cp_wait0(); }
    __syncthreads();                               // tile kt visible to all waves

    const _Float16* as = SA[kt & 1];
    const _Float16* bs = SB[kt & 1];
    h16 af[2], bf[4];
    #pragma unroll
    for (int i = 0; i < 2; ++i) {
      const _Float16* base = as + (wm * 32 + i * 16 + rl) * 32;
      h8 a0 = *(const h8*)(base + hi * 8);         // K = 8*hi + 0..7
      h8 a1 = *(const h8*)(base + 16 + hi * 8);    // K = 16 + 8*hi + 0..7
      h16 t;
      #pragma unroll
      for (int u = 0; u < 8; ++u) { t[u] = a0[u]; t[8 + u] = a1[u]; }
      af[i] = t;
    }
    #pragma unroll
    for (int j = 0; j < 4; ++j) {
      const _Float16* base = bs + (wn * 64 + j * 16 + rl) * 32;
      h8 b0 = *(const h8*)(base + hi * 8);         // K = 8*hi + 0..7
      h8 b1 = *(const h8*)(base + 16 + hi * 8);    // K = 16 + 8*hi + 0..7
      h16 t;
      #pragma unroll
      for (int u = 0; u < 8; ++u) { t[u] = b0[u]; t[8 + u] = b1[u]; }
      bf[j] = t;
    }
    #pragma unroll
    for (int i = 0; i < 2; ++i)
      #pragma unroll
      for (int j = 0; j < 4; ++j)
        acc[i][j] = __builtin_amdgcn_wmma_f32_16x16x32_f16(
            false, af[i], false, bf[j], (short)0, acc[i][j], false, false);

    __syncthreads();                               // all waves done with buffer kt&1
  }

  // ---- epilogue ----
  #pragma unroll
  for (int j = 0; j < 4; ++j) {
    int col = n0 + wn * 64 + j * 16 + rl;
    if (col >= N) continue;
    float bv = bias ? bias[col] : 0.f;
    #pragma unroll
    for (int i = 0; i < 2; ++i) {
      #pragma unroll
      for (int r = 0; r < 8; ++r) {
        int row = m0 + wm * 32 + i * 16 + hi * 8 + r;
        if (row >= M) continue;
        float v = alpha * acc[i][j][r] + bv;
        if (out_mode == 0)      ((float*)Cv + (long)blockIdx.z * sC)[(long)row * ldc + col] = v;
        else if (out_mode == 1) ((_Float16*)Cv + (long)blockIdx.z * sC)[(long)row * ldc + col] = (_Float16)v;
        else                    ((_Float16*)Cv + (long)blockIdx.z * sC)[(long)col * ldc + row] = (_Float16)v;
      }
    }
  }
}

// ---------------- conversion / elementwise kernels ----------------
__global__ void camil_fill(float* p, float v, long n) {
  long i = (long)blockIdx.x * 256 + threadIdx.x;
  if (i < n) p[i] = v;
}

// bag -> f16 padded [NPD+128, DD] (front pad + tail pad rows zeroed)
__global__ void camil_pad_h(const float* __restrict__ bag, _Float16* __restrict__ xph) {
  long i = (long)blockIdx.x * 256 + threadIdx.x;
  if (i >= (long)(NPD + 128) * DD) return;
  long r = i >> 10;
  float v = 0.f;
  if (r >= PADR && r < NPD) v = bag[i - (long)PADR * DD];
  xph[i] = (_Float16)v;
}

// weight f32 [K,N] -> f16 [N,K]
__global__ void camil_wtrans(const float* __restrict__ in, _Float16* __restrict__ out,
                             int K, int N) {
  long i = (long)blockIdx.x * 256 + threadIdx.x;
  if (i >= (long)N * K) return;
  int n = (int)(i / K), k = (int)(i % K);
  out[i] = (_Float16)in[(long)k * N + n];
}

// v^T: VTh[h][d][n] = qkvh[n][1024 + h*64 + d]
__global__ void camil_vtrans(const _Float16* __restrict__ qkvh, _Float16* __restrict__ vt) {
  long i = (long)blockIdx.x * 256 + threadIdx.x;
  if (i >= (long)NH * DHD * NPD) return;
  int n = (int)(i % NPD);
  long hd = i / NPD;               // h*64 + d
  vt[i] = qkvh[(long)n * QKD + 1024 + hd];
}

// landmark means: ql (with 1/sqrt(DH) baked in) and kl, f16 out
__global__ void camil_landmarks_h(const _Float16* __restrict__ qkvh,
                                  _Float16* __restrict__ ql, _Float16* __restrict__ kl) {
  int m = blockIdx.x & 255, h = blockIdx.x >> 8;
  int d = threadIdx.x;                    // 64 threads
  const _Float16* base = qkvh + (long)(m * LSEG) * QKD + h * DHD + d;
  float sq = 0.f, sk = 0.f;
  for (int j = 0; j < LSEG; ++j) {
    sq += (float)base[(long)j * QKD];
    sk += (float)base[(long)j * QKD + 512];
  }
  ql[((long)h * MLM + m) * DHD + d] = (_Float16)(sq * (0.125f / (float)LSEG));
  kl[((long)h * MLM + m) * DHD + d] = (_Float16)(sk * (1.0f / (float)LSEG));
}

// in-place f16 row softmax
__global__ __launch_bounds__(256)
void camil_softmax_rows_h(_Float16* __restrict__ X, int cols) {
  __shared__ float sm[256];
  _Float16* x = X + (long)blockIdx.x * cols;
  float m = -3.4e38f;
  for (int c = threadIdx.x; c < cols; c += 256) m = fmaxf(m, (float)x[c]);
  m = camil_blk_max(m, sm);
  float s = 0.f;
  for (int c = threadIdx.x; c < cols; c += 256) s += __expf((float)x[c] - m);
  s = camil_blk_sum(s, sm);
  float inv = 1.f / s;
  for (int c = threadIdx.x; c < cols; c += 256) x[c] = (_Float16)(__expf((float)x[c] - m) * inv);
}

__global__ __launch_bounds__(256)
void camil_softmax1d(const float* __restrict__ x, float* __restrict__ y, int n) {
  __shared__ float sm[256];
  float m = -3.4e38f;
  for (int i = threadIdx.x; i < n; i += 256) m = fmaxf(m, x[i]);
  m = camil_blk_max(m, sm);
  float s = 0.f;
  for (int i = threadIdx.x; i < n; i += 256) s += __expf(x[i] - m);
  s = camil_blk_sum(s, sm);
  float inv = 1.f / s;
  for (int i = threadIdx.x; i < n; i += 256) y[i] = __expf(x[i] - m) * inv;
}

// global max of column sums of softmaxed a2 (row-sums are exactly 1)
__global__ __launch_bounds__(256)
void camil_colmax_h(const _Float16* __restrict__ a2, float* __restrict__ gmax) {
  __shared__ float sm[256];
  const _Float16* x = a2 + (long)blockIdx.x * MLM * MLM;
  int j = threadIdx.x;
  float s = 0.f;
  for (int i = 0; i < MLM; ++i) s += (float)x[(long)i * MLM + j];
  float m = camil_blk_max(s, sm);
  if (threadIdx.x == 0) atomicMax(reinterpret_cast<int*>(gmax), __float_as_int(m));
}

// z0 = a2^T / max_colsum : writes Zh (row-major) and ZTh (=a2/s, the B-layout)
__global__ void camil_zinit_h(const _Float16* __restrict__ a2, const float* __restrict__ gmax,
                              _Float16* __restrict__ zh, _Float16* __restrict__ zth) {
  long i = (long)blockIdx.x * 256 + threadIdx.x;
  if (i >= (long)NH * MLM * MLM) return;
  long b = i >> 16; int r = (int)((i >> 8) & 255), c = (int)(i & 255);
  float v = (float)a2[i] / gmax[0];
  zth[i] = (_Float16)v;                            // ZTh[b][r][c] = a2[b][r][c]/s
  zh[(b << 16) + (long)c * MLM + r] = (_Float16)v; // Zh[b][c][r]  = a2[b][r][c]/s
}

// WTh = (a*I - X)^T = a*I - X^T   (f16 in row-major, f16 out B-layout)
__global__ void camil_ai_minus_t(const _Float16* __restrict__ x, _Float16* __restrict__ y, float a) {
  long i = (long)blockIdx.x * 256 + threadIdx.x;
  if (i >= (long)NH * MLM * MLM) return;
  long b = i >> 16; int j = (int)((i >> 8) & 255), k = (int)(i & 255);
  float v = (j == k ? a : 0.f) - (float)x[(b << 16) + (long)k * MLM + j];
  y[i] = (_Float16)v;
}

// f16 batched 256x256 transpose (new z -> B-layout)
__global__ void camil_htrans(const _Float16* __restrict__ in, _Float16* __restrict__ out) {
  long i = (long)blockIdx.x * 256 + threadIdx.x;
  if (i >= (long)NH * MLM * MLM) return;
  long b = i >> 16; int r = (int)((i >> 8) & 255), c = (int)(i & 255);
  out[i] = in[(b << 16) + (long)c * MLM + r];
}

// depthwise residual conv along sequence; attn f32 + conv -> f16 [NPD,512]
__global__ void camil_conv_add_h(const _Float16* __restrict__ qkvh, const float* __restrict__ rw,
                                 const float* __restrict__ attn, _Float16* __restrict__ atth) {
  long i = (long)blockIdx.x * 256 + threadIdx.x;
  if (i >= (long)NPD * (NH * DHD)) return;
  int n = (int)(i >> 9), c = (int)(i & 511);
  int h = c >> 6, d = c & 63;
  float s = attn[i];
  #pragma unroll
  for (int j = 0; j < KCV; ++j) {
    int nn = n + j - (KCV / 2);
    if (nn >= 0 && nn < NPD) s += (float)qkvh[(long)nn * QKD + 1024 + h * DHD + d] * rw[h * KCV + j];
  }
  atth[i] = (_Float16)s;
}

// encoder_output = enc[PADR:] + bag (f32 + f16 copies)
__global__ void camil_resid2(const float* __restrict__ enc, const float* __restrict__ bag,
                             float* __restrict__ eno, _Float16* __restrict__ enoh) {
  long i = (long)blockIdx.x * 256 + threadIdx.x;
  if (i >= (long)NI * DD) return;
  float v = enc[i + (long)PADR * DD] + bag[i];
  eno[i] = v;
  enoh[i] = (_Float16)v;
}

// per-edge dot product + segment sum (one wave per edge)
__global__ __launch_bounds__(256)
void camil_edges(const float* __restrict__ qc, const float* __restrict__ kc,
                 const float* __restrict__ vals, const int* __restrict__ rows,
                 const int* __restrict__ cols, float* __restrict__ araw) {
  int e = blockIdx.x * 8 + (threadIdx.x >> 5);
  if (e >= NE) return;
  int lane = threadIdx.x & 31;
  int r = rows[e], c = cols[e];
  const float* qp = qc + (long)r * 256;
  const float* kp = kc + (long)c * 256;
  float s = 0.f;
  #pragma unroll
  for (int t = 0; t < 8; ++t) s += qp[lane + 32 * t] * kp[lane + 32 * t];
  #pragma unroll
  for (int o = 16; o > 0; o >>= 1) s += __shfl_xor(s, o, 32);
  if (lane == 0) atomicAdd(araw + r, s * vals[e] * 0.0625f);   // 1/sqrt(256)
}

// xo = xl*2*sw + 2*encO*(1-sw),  xl = alpha[row]*value, sw = sigmoid(-xl)^2
__global__ void camil_gate(const float* __restrict__ value, const float* __restrict__ eno,
                           const float* __restrict__ alpha, float* __restrict__ xo) {
  long i = (long)blockIdx.x * 256 + threadIdx.x;
  if (i >= (long)NI * DD) return;
  int row = (int)(i >> 10);
  float xl = alpha[row] * value[i];
  float sg = 1.f / (1.f + __expf(xl));          // sigmoid(-xl)
  float sw = sg * sg;
  xo[i] = xl * 2.f * sw + 2.f * eno[i] * (1.f - sw);
}

// pooled[d] = mean_i xo[i][d]  (k_alpha == 1: softmax over singleton axis)
__global__ __launch_bounds__(256)
void camil_pool(const float* __restrict__ xo, float* __restrict__ pooled) {
  __shared__ float sm[256];
  int d = blockIdx.x;
  float s = 0.f;
  for (int i = threadIdx.x; i < NI; i += 256) s += xo[(long)i * DD + d];
  s = camil_blk_sum(s, sm);
  if (threadIdx.x == 0) pooled[d] = s * (1.f / (float)NI);
}

__global__ __launch_bounds__(256)
void camil_fc(const float* __restrict__ pooled, const float* __restrict__ fcw,
              const float* __restrict__ fcb, const float* __restrict__ fcb2,
              float* __restrict__ out) {
  __shared__ float sm[256];
  for (int c = 0; c < 2; ++c) {
    float s = 0.f;
    for (int d = threadIdx.x; d < DD; d += 256) s += pooled[d] * fcw[d * 2 + c];
    s = camil_blk_sum(s, sm);
    if (threadIdx.x == 0) out[c] = s + fcb[c] + fcb2[c];
    __syncthreads();
  }
}

// ---------------- host driver ----------------
extern "C" void kernel_launch(void* const* d_in, const int* in_sizes, int n_in,
                              void* d_out, int out_size, void* d_ws, size_t ws_size,
                              hipStream_t stream) {
  const float* bag      = (const float*)d_in[0];
  const float* adj_vals = (const float*)d_in[1];
  const float* qkv_w    = (const float*)d_in[2];
  const float* out_w    = (const float*)d_in[3];
  const float* out_b    = (const float*)d_in[4];
  const float* res_w    = (const float*)d_in[5];
  const float* wq_w     = (const float*)d_in[6];
  const float* wq_b     = (const float*)d_in[7];
  const float* wk_w     = (const float*)d_in[8];
  const float* wk_b     = (const float*)d_in[9];
  const float* wv_w     = (const float*)d_in[10];
  const float* wv_b     = (const float*)d_in[11];
  // d_in[12..17] (va/ua/wa): dead code — k_alpha = softmax over singleton axis == 1
  const float* fc_w     = (const float*)d_in[18];
  const float* fc_b     = (const float*)d_in[19];
  const float* fc_b2    = (const float*)d_in[20];
  const int*   adj_rows = (const int*)d_in[21];
  const int*   adj_cols = (const int*)d_in[22];

  float* OUT  = (float*)d_out;
  float* ARAW = OUT + 2;           // [NI]
  float* KALP = OUT + 2 + NI;      // [NI]

  // ---- workspace: f32 region then f16 region (all 128-row padded for async) ----
  float* Wf = (float*)d_ws;
  size_t of = 0;
  auto af32 = [&](size_t n) { float* p = Wf + of; of += (n + 63) & ~(size_t)63; return p; };
  float* ATT  = af32((size_t)NPD * 512);
  float* ENC  = af32((size_t)NPD * DD);
  float* ENO  = af32((size_t)NI * DD);
  float* QC   = af32((size_t)NI * 256);
  float* KC_  = af32((size_t)NI * 256);
  float* VAL  = af32((size_t)NI * DD);
  float* XO   = af32((size_t)NI * DD);
  float* ALP  = af32(NI);
  float* GMX  = af32(64);
  float* POOL = af32(DD);

  _Float16* Wh = (_Float16*)(Wf + of);
  size_t oh = 0;
  auto af16 = [&](size_t n) { _Float16* p = Wh + oh; oh += (n + 63) & ~(size_t)63; return p; };
  _Float16* XPh   = af16((size_t)(NPD + 128) * DD);
  _Float16* QKVh  = af16((size_t)NPD * QKD);
  _Float16* WQKVT = af16((size_t)QKD * DD);
  _Float16* WOUTT = af16((size_t)DD * 512);
  _Float16* WWQT  = af16((size_t)256 * DD);
  _Float16* WWKT  = af16((size_t)256 * DD);
  _Float16* WWVT  = af16((size_t)DD * DD);
  _Float16* QLh   = af16((size_t)NH * MLM * DHD);
  _Float16* KLh   = af16((size_t)NH * MLM * DHD);
  _Float16* A1h   = af16((size_t)NH * NPD * MLM);
  _Float16* A2h   = af16((size_t)NH * MLM * MLM);
  _Float16* A3h   = af16((size_t)NH * MLM * NPD);
  _Float16* VTh   = af16((size_t)(NH * DHD + 64) * NPD);   // +64 pad rows (B staging)
  _Float16* Zh    = af16((size_t)NH * MLM * MLM);
  _Float16* ZTh   = af16((size_t)NH * MLM * MLM);
  _Float16* XZh   = af16((size_t)NH * MLM * MLM);
  _Float16* WTh   = af16((size_t)NH * MLM * MLM);
  _Float16* W2h   = af16((size_t)NH * MLM * MLM);
  _Float16* ZNh   = af16((size_t)NH * MLM * MLM);
  _Float16* A3VTh = af16((size_t)(NH * DHD + 64) * MLM);   // +64 pad rows
  _Float16* PWTh  = af16((size_t)(NH * DHD + 64) * MLM);   // +64 pad rows
  _Float16* ATTh  = af16((size_t)NPD * 512);
  _Float16* ENOh  = af16((size_t)NIP * DD);                // 20096 rows

  auto gemm = [&](const _Float16* A, const _Float16* B, void* C, const float* bias,
                  int M, int N, int K, int lda, int ldb, int ldc,
                  long sA, long sB, long sC, int batch, float alpha, int mode) {
    dim3 g((N + 127) / 128, (M + 127) / 128, batch);
    camil_gemm_wmma_h<<<g, dim3(256), 0, stream>>>(A, B, C, bias, M, N, K,
                                                   lda, ldb, ldc, sA, sB, sC, alpha, mode);
  };
  auto ew = [&](long n) { return dim3((unsigned)((n + 255) / 256)); };

  const long s2  = (long)MLM * MLM;
  const long sLM = (long)MLM * DHD;

  // init outputs / scalars
  camil_fill<<<ew(NI), 256, 0, stream>>>(ARAW, 0.f, NI);
  camil_fill<<<ew(NI), 256, 0, stream>>>(KALP, 1.f, NI);   // softmax over size-1 axis
  camil_fill<<<1, 256, 0, stream>>>(GMX, 0.f, 64);

  // ---- one-time f16 operand preparation ----
  camil_pad_h<<<ew((long)(NPD + 128) * DD), 256, 0, stream>>>(bag, XPh);
  camil_wtrans<<<ew((long)QKD * DD), 256, 0, stream>>>(qkv_w, WQKVT, DD, QKD);
  camil_wtrans<<<ew((long)DD * 512), 256, 0, stream>>>(out_w, WOUTT, 512, DD);
  camil_wtrans<<<ew((long)256 * DD), 256, 0, stream>>>(wq_w, WWQT, DD, 256);
  camil_wtrans<<<ew((long)256 * DD), 256, 0, stream>>>(wk_w, WWKT, DD, 256);
  camil_wtrans<<<ew((long)DD * DD), 256, 0, stream>>>(wv_w, WWVT, DD, DD);

  // ---- Nystrom attention ----
  gemm(XPh, WQKVT, QKVh, nullptr, NPD, QKD, DD, DD, DD, QKD, 0, 0, 0, 1, 1.f, 1);
  camil_landmarks_h<<<NH * MLM, DHD, 0, stream>>>(QKVh, QLh, KLh);
  camil_vtrans<<<ew((long)NH * DHD * NPD), 256, 0, stream>>>(QKVh, VTh);

  // a1 = softmax( 0.125 * q . kl^T )
  gemm(QKVh, KLh, A1h, nullptr, NPD, MLM, DHD, QKD, DHD, MLM,
       64, sLM, (long)NPD * MLM, NH, 0.125f, 1);
  camil_softmax_rows_h<<<NH * NPD, 256, 0, stream>>>(A1h, MLM);
  // a2 = softmax( ql . kl^T )   (ql already scaled)
  gemm(QLh, KLh, A2h, nullptr, MLM, MLM, DHD, DHD, DHD, MLM, sLM, sLM, s2, NH, 1.f, 1);
  camil_softmax_rows_h<<<NH * MLM, 256, 0, stream>>>(A2h, MLM);
  // a3 = softmax( ql . k^T )
  gemm(QLh, QKVh + 512, A3h, nullptr, MLM, NPD, DHD, DHD, QKD, NPD,
       sLM, 64, (long)MLM * NPD, NH, 1.f, 1);
  camil_softmax_rows_h<<<NH * MLM, 256, 0, stream>>>(A3h, NPD);

  // pinv(a2) via iterative scheme; operands kept in both layouts as needed
  camil_colmax_h<<<NH, 256, 0, stream>>>(A2h, GMX);
  camil_zinit_h<<<ew(NH * s2), 256, 0, stream>>>(A2h, GMX, Zh, ZTh);
  for (int it = 0; it < 6; ++it) {
    gemm(A2h, ZTh, XZh, nullptr, MLM, MLM, MLM, MLM, MLM, MLM, s2, s2, s2, NH, 1.f, 1);
    camil_ai_minus_t<<<ew(NH * s2), 256, 0, stream>>>(XZh, WTh, 7.f);
    gemm(XZh, WTh, W2h, nullptr, MLM, MLM, MLM, MLM, MLM, MLM, s2, s2, s2, NH, 1.f, 1);
    camil_ai_minus_t<<<ew(NH * s2), 256, 0, stream>>>(W2h, WTh, 15.f);
    gemm(XZh, WTh, W2h, nullptr, MLM, MLM, MLM, MLM, MLM, MLM, s2, s2, s2, NH, 1.f, 1);
    camil_ai_minus_t<<<ew(NH * s2), 256, 0, stream>>>(W2h, WTh, 13.f);
    gemm(Zh, WTh, ZNh, nullptr, MLM, MLM, MLM, MLM, MLM, MLM, s2, s2, s2, NH, 0.25f, 1);
    camil_htrans<<<ew(NH * s2), 256, 0, stream>>>(ZNh, ZTh);
    _Float16* t = Zh; Zh = ZNh; ZNh = t;
  }

  // a3v^T, pw^T, attn  (re-associated (a1@pinv)@(a3@v) = a1@(pinv@(a3@v)))
  gemm(A3h, VTh, A3VTh, nullptr, MLM, DHD, NPD, NPD, NPD, MLM,
       (long)MLM * NPD, (long)DHD * NPD, (long)DHD * MLM, NH, 1.f, 2);
  gemm(Zh, A3VTh, PWTh, nullptr, MLM, DHD, MLM, MLM, MLM, MLM,
       s2, (long)DHD * MLM, (long)DHD * MLM, NH, 1.f, 2);
  gemm(A1h, PWTh, ATT, nullptr, NPD, DHD, MLM, MLM, MLM, NH * DHD,
       (long)NPD * MLM, (long)DHD * MLM, 64, NH, 1.f, 0);
  camil_conv_add_h<<<ew((long)NPD * NH * DHD), 256, 0, stream>>>(QKVh, res_w, ATT, ATTh);

  // enc = attn @ out_w + out_b ; encoder_output = enc[PADR:] + bag
  gemm(ATTh, WOUTT, ENC, out_b, NPD, DD, NH * DHD, NH * DHD, NH * DHD, DD, 0, 0, 0, 1, 1.f, 0);
  camil_resid2<<<ew((long)NI * DD), 256, 0, stream>>>(ENC, bag, ENO, ENOh);

  // ---- custom sparse attention ----
  gemm(ENOh, WWQT, QC, wq_b, NI, 256, DD, DD, DD, 256, 0, 0, 0, 1, 1.f, 0);
  gemm(ENOh, WWKT, KC_, wk_b, NI, 256, DD, DD, DD, 256, 0, 0, 0, 1, 1.f, 0);
  camil_edges<<<(NE + 7) / 8, 256, 0, stream>>>(QC, KC_, adj_vals, adj_rows, adj_cols, ARAW);
  camil_softmax1d<<<1, 256, 0, stream>>>(ARAW, ALP, NI);

  // value = bag @ wv_w + wv_b (A view: rows PADR.. of XPh); gated combine
  gemm(XPh + (size_t)PADR * DD, WWVT, VAL, wv_b, NI, DD, DD, DD, DD, DD, 0, 0, 0, 1, 1.f, 0);
  camil_gate<<<ew((long)NI * DD), 256, 0, stream>>>(VAL, ENO, ALP, XO);

  // ---- pooling + classifier (MIL attention branch dead: k_alpha == 1) ----
  camil_pool<<<DD, 256, 0, stream>>>(XO, POOL);
  camil_fc<<<1, 256, 0, stream>>>(POOL, fc_w, fc_b, fc_b2, OUT);
}
